// Gating_40424232190280
// MI455X (gfx1250) — matrix operand order repaired and checked by reading, not compiled, not run on hardware
//
#include <hip/hip_runtime.h>
#include <hip/hip_bf16.h>

// CDNA5 WMMA vector types (probe-confirmed signatures)
typedef __attribute__((ext_vector_type(16))) __bf16 v16bf;
typedef __attribute__((ext_vector_type(8)))  float  v8f;

#define TOKENS   16384
#define DIM      2048
#define NEXP     64
#define KC       128          // K-chunk staged in LDS
#define KCP      (KC + 8)     // padded row (272 B) -> bank-conflict friendly, 16B-aligned rows
#define BLK_TOK  128          // tokens per block (8 waves x 16 tokens)

__global__ __launch_bounds__(256, 1)
void gating_topk_kernel(const float* __restrict__ x,
                        const float* __restrict__ Wg,
                        float* __restrict__ out)
{
    // W chunk as bf16 hi/lo split: 2 x 64 x 136 x 2B = 34.8 KB LDS
    __shared__ alignas(16) __bf16 whi[NEXP][KCP];
    __shared__ alignas(16) __bf16 wlo[NEXP][KCP];

    const int tid  = threadIdx.x;
    const int wave = tid >> 5;
    const int lane = tid & 31;
    const int half = lane >> 4;   // 0: lanes 0-15, 1: lanes 16-31
    const int l16  = lane & 15;

    const int tokBase = blockIdx.x * BLK_TOK + wave * 16;
    const float* xrow = x + (size_t)(tokBase + l16) * DIM;   // A row M = l16

    // 4 accumulator tiles: experts [0..15],[16..31],[32..47],[48..63]
    v8f acc[4];
    #pragma unroll
    for (int t = 0; t < 4; ++t)
        #pragma unroll
        for (int i = 0; i < 8; ++i) acc[t][i] = 0.0f;

    // cooperative W-chunk conversion mapping: thread -> (expert, 32-wide k quarter)
    const int we = tid >> 2;
    const int wq = (tid & 3) * 32;
    const float* wrowBase = Wg + (size_t)we * DIM + wq;

    for (int kc = 0; kc < DIM; kc += KC) {
        __syncthreads();   // previous chunk fully consumed before overwrite
        // ---- stage W[64][KC] as bf16 hi + bf16 lo residual into LDS ----
        #pragma unroll
        for (int j = 0; j < 8; ++j) {
            float4 w = *(const float4*)(wrowBase + kc + j * 4);
            union { __bf16 e[4]; uint2 q; } uh, ul;
            #pragma unroll
            for (int c = 0; c < 4; ++c) {
                float f = (&w.x)[c];
                __bf16 h = (__bf16)f;            // RNE truncate to bf16
                uh.e[c] = h;
                ul.e[c] = (__bf16)(f - (float)h); // bf16 residual
            }
            *(uint2*)&whi[we][wq + j * 4] = uh.q;
            *(uint2*)&wlo[we][wq + j * 4] = ul.q;
        }
        __syncthreads();

        if (kc + KC < DIM)                        // warm next x chunk (global_prefetch_b8)
            __builtin_prefetch(xrow + kc + KC, 0, 3);

        // ---- K-steps of 32 within the chunk ----
        #pragma unroll
        for (int s = 0; s < KC; s += 32) {
            const int k0  = kc + s;
            const int seg = half * 16; // A uses contiguous k order, same as B: lanes<16 hold K 0-15; lanes>=16 hold K 16-31
            float4 q0 = *(const float4*)(xrow + k0 + seg);
            float4 q1 = *(const float4*)(xrow + k0 + seg + 4);
            float4 q2 = *(const float4*)(xrow + k0 + seg + 8);
            float4 q3 = *(const float4*)(xrow + k0 + seg + 12);
            float f[16] = { q0.x,q0.y,q0.z,q0.w, q1.x,q1.y,q1.z,q1.w,
                            q2.x,q2.y,q2.z,q2.w, q3.x,q3.y,q3.z,q3.w };
            v16bf ahi, alo;
            #pragma unroll
            for (int i = 0; i < 16; ++i) {
                __bf16 h = (__bf16)f[i];
                ahi[i] = h;
                alo[i] = (__bf16)(f[i] - (float)h);
            }

            // B layout: lane n holds column (expert) n%16, K = step + (n>=16 ? 16 : 0) .. +15
            const int kl = s + half * 16;
            #pragma unroll
            for (int t = 0; t < 4; ++t) {
                const __bf16* ph = &whi[t * 16 + l16][kl];
                const __bf16* pl = &wlo[t * 16 + l16][kl];
                union { v16bf v; uint4 q[2]; } ub, vb;   // 2x ds_load_b128 each
                ub.q[0] = ((const uint4*)ph)[0];
                ub.q[1] = ((const uint4*)ph)[1];
                vb.q[0] = ((const uint4*)pl)[0];
                vb.q[1] = ((const uint4*)pl)[1];
                // fp32-accurate product via 3-term bf16 split: hi*hi + lo*hi + hi*lo
                acc[t] = __builtin_amdgcn_wmma_f32_16x16x32_bf16(false, ahi, false, ub.v, (short)0, acc[t], false, false);
                acc[t] = __builtin_amdgcn_wmma_f32_16x16x32_bf16(false, alo, false, ub.v, (short)0, acc[t], false, false);
                acc[t] = __builtin_amdgcn_wmma_f32_16x16x32_bf16(false, ahi, false, vb.v, (short)0, acc[t], false, false);
            }
        }
    }

    // ---- fused top-2 + softmax ----
    // C layout: VGPR v, lanes 0-15 -> (token v, expert lane), lanes 16-31 -> (token v+8, expert lane-16).
    // For token row v, the 64 expert logits live across one 16-lane group, 4 per lane (one per tile).
    #pragma unroll
    for (int v = 0; v < 8; ++v) {
        float a0 = acc[0][v], a1 = acc[1][v], a2 = acc[2][v], a3 = acc[3][v];
        float h01 = fmaxf(a0, a1), l01 = fminf(a0, a1);
        float h23 = fmaxf(a2, a3), l23 = fminf(a2, a3);
        float m1 = fmaxf(h01, h23);
        float m2 = fmaxf(fminf(h01, h23), fmaxf(l01, l23));
        // butterfly max/second-max across the 16-lane group (xor<16 stays in-group)
        #pragma unroll
        for (int msk = 1; msk < 16; msk <<= 1) {
            float o1 = __shfl_xor(m1, msk, 32);
            float o2 = __shfl_xor(m2, msk, 32);
            float nm1 = fmaxf(m1, o1);
            float nm2 = fmaxf(fminf(m1, o1), fmaxf(m2, o2));
            m1 = nm1; m2 = nm2;
        }
        if (l16 == 0) {
            float e   = __expf(m2 - m1);      // <= 1, numerically safe
            float inv = 1.0f / (1.0f + e);
            int tok = tokBase + v + half * 8;
            out[2 * tok]     = inv;           // softmax of sorted-descending top-2
            out[2 * tok + 1] = e * inv;
        }
    }
}

extern "C" void kernel_launch(void* const* d_in, const int* in_sizes, int n_in,
                              void* d_out, int out_size, void* d_ws, size_t ws_size,
                              hipStream_t stream) {
    (void)in_sizes; (void)n_in; (void)d_ws; (void)ws_size; (void)out_size;
    const float* x  = (const float*)d_in[0];   // [TOKENS, DIM] f32
    const float* Wg = (const float*)d_in[1];   // [NEXP, DIM] f32
    float* out = (float*)d_out;                // [TOKENS, 2] f32
    dim3 grid(TOKENS / BLK_TOK);
    gating_topk_kernel<<<grid, 256, 0, stream>>>(x, Wg, out);
}
